// Slayer_grouped_16655883173897
// MI455X (gfx1250) — matrix operand order repaired and checked by reading, not compiled, not run on hardware
//
#include <hip/hip_runtime.h>
#include <hip/hip_bf16.h>

// Problem constants (match reference)
#define BATCH   256
#define IN_DIM  256
#define OUT_DIM 256
#define NUM_F   8
#define KTOT    (IN_DIM * (NUM_F + 1))   // 2304 = 72 * 32

typedef __attribute__((ext_vector_type(16))) _Float16 v16h;
typedef __attribute__((ext_vector_type(8)))  _Float16 v8h;
typedef __attribute__((ext_vector_type(8)))  float    v8f;

// ---------------------------------------------------------------------------
// Phase 1a: F[b][f*256+i] = { silu(x[b,i]) , sin(grid[k]*x[b,i]) } as f16.
// f-major K layout -> stores are coalesced along i across the wave.
// ---------------------------------------------------------------------------
__global__ void slayer_build_F(const float* __restrict__ x,
                               const float* __restrict__ grid,
                               _Float16* __restrict__ F) {
  int idx = blockIdx.x * blockDim.x + threadIdx.x;   // b*IN_DIM + i
  int b = idx >> 8;
  int i = idx & (IN_DIM - 1);
  float xv = x[idx];
  float silu = xv / (1.0f + __expf(-xv));
  _Float16* row = F + (size_t)b * KTOT;
  row[i] = (_Float16)silu;
#pragma unroll
  for (int k = 0; k < NUM_F; ++k) {
    row[(k + 1) * IN_DIM + i] = (_Float16)__sinf(grid[k] * xv);
  }
}

// ---------------------------------------------------------------------------
// Phase 1b: W[o][f*256+i] = { scale_base[o,i] , scale_sp[o,i]*coef[o/4][k] }.
// ---------------------------------------------------------------------------
__global__ void slayer_build_W(const float* __restrict__ coef,
                               const float* __restrict__ scale_base,
                               const float* __restrict__ scale_sp,
                               _Float16* __restrict__ W) {
  int idx = blockIdx.x * blockDim.x + threadIdx.x;   // o*IN_DIM + i
  int o = idx >> 8;
  int i = idx & (IN_DIM - 1);
  int g = o >> 2;                 // group = (o*256+i)/1024 = o/4
  float sb = scale_base[idx];
  float sp = scale_sp[idx];
  _Float16* row = W + (size_t)o * KTOT;
  row[i] = (_Float16)sb;
#pragma unroll
  for (int k = 0; k < NUM_F; ++k) {
    row[(k + 1) * IN_DIM + i] = (_Float16)(sp * coef[g * NUM_F + k]);
  }
}

// ---------------------------------------------------------------------------
// Phase 2: y = F (256 x 2304) @ W^T (2304 x 256) + bias, via
// v_wmma_f32_16x16x32_f16. One wave per 16x16 output tile; fragments are
// loaded straight from global memory with b128 loads matching the ISA's
// A/B VGPR layouts (rows are 4608 B, all slices 16B-aligned).
// ---------------------------------------------------------------------------
__global__ void __launch_bounds__(128)
slayer_wmma_gemm(const _Float16* __restrict__ F,
                 const _Float16* __restrict__ W,
                 const float* __restrict__ bias_w,
                 float* __restrict__ y) {
  const int wave = (blockIdx.x * blockDim.x + threadIdx.x) >> 5;  // 0..255
  const int lane = threadIdx.x & 31;
  const int tm   = wave >> 4;          // batch tile 0..15
  const int tn   = wave & 15;          // out   tile 0..15
  const int half = lane >> 4;          // 0: lanes 0-15, 1: lanes 16-31
  const int l    = lane & 15;

  // A: lane holds row (tm*16 + l); K halves at +0/+16 (lanes<16) or +8/+24.
  const _Float16* Frow = F + (size_t)(tm * 16 + l) * KTOT;
  // B: lane holds col (tn*16 + l); K halves at +0/+16 (lanes<16) or +8/+24.
  const int ncol = tn * 16 + l;
  const _Float16* Wrow = W + (size_t)ncol * KTOT;

  const int a_off = half * 8;    // elements
  const int b_off = half * 8;

  // C/D layout: VGPR r -> row r + 8*half, col (lane&15). Bias is per-column,
  // identical across all 8 accumulator VGPRs.
  const float bv = bias_w[ncol];
  v8f c;
#pragma unroll
  for (int r = 0; r < 8; ++r) c[r] = bv;

#pragma unroll 4
  for (int kb = 0; kb < KTOT; kb += 32) {
    v8h a_lo = *(const v8h*)(Frow + kb + a_off);        // K a_off+0..7
    v8h a_hi = *(const v8h*)(Frow + kb + a_off + 16);   // K a_off+16..23
    v8h b_lo = *(const v8h*)(Wrow + kb + b_off);        // K b_off+0..7
    v8h b_hi = *(const v8h*)(Wrow + kb + b_off + 16);   // K b_off+16..23
    v16h A = __builtin_shufflevector(a_lo, a_hi,
                                     0,1,2,3,4,5,6,7,8,9,10,11,12,13,14,15);
    v16h B = __builtin_shufflevector(b_lo, b_hi,
                                     0,1,2,3,4,5,6,7,8,9,10,11,12,13,14,15);
    c = __builtin_amdgcn_wmma_f32_16x16x32_f16(
        /*neg_a=*/false, A, /*neg_b=*/false, B,
        /*c_mod=*/(short)0, c, /*reuse_a=*/false, /*reuse_b=*/false);
  }

#pragma unroll
  for (int r = 0; r < 8; ++r) {
    int row = tm * 16 + half * 8 + r;
    y[row * OUT_DIM + ncol] = c[r];
  }
}

// ---------------------------------------------------------------------------
// Harness entry. Inputs (setup_inputs order):
//   0: x (256*256 f32), 1: grid (8 f32), 2: coef (64*1*8 f32),
//   3: bias_w (1*256 f32), 4: scale_base (65536 f32), 5: scale_sp (65536 f32)
// Output: y (256*256 f32).
// Workspace layout: F f16 [256][2304] then W f16 [256][2304] (~2.36 MB).
// ---------------------------------------------------------------------------
extern "C" void kernel_launch(void* const* d_in, const int* in_sizes, int n_in,
                              void* d_out, int out_size, void* d_ws, size_t ws_size,
                              hipStream_t stream) {
  const float* x          = (const float*)d_in[0];
  const float* grid       = (const float*)d_in[1];
  const float* coef       = (const float*)d_in[2];
  const float* bias_w     = (const float*)d_in[3];
  const float* scale_base = (const float*)d_in[4];
  const float* scale_sp   = (const float*)d_in[5];
  float* y = (float*)d_out;

  _Float16* F = (_Float16*)d_ws;
  _Float16* W = F + (size_t)BATCH * KTOT;

  slayer_build_F<<<BATCH, 256, 0, stream>>>(x, grid, F);
  slayer_build_W<<<OUT_DIM, 256, 0, stream>>>(coef, scale_base, scale_sp, W);
  slayer_wmma_gemm<<<64, 128, 0, stream>>>(F, W, bias_w, y);
}
